// BottleNeck_6949257085183
// MI455X (gfx1250) — 8-bit matrix operand order repaired and checked by reading, compiled, not run on hardware
//
#include <hip/hip_runtime.h>
#include <cstdint>


typedef __attribute__((ext_vector_type(8))) int v8i;

#define EPS 1e-5f

__device__ __forceinline__ int sgnb(float x) {
    return (x <= 0.0f) ? 0xFF : 0x01;
}
__device__ __forceinline__ int pack4(int b0, int b1, int b2, int b3) {
    return (b0 & 0xFF) | ((b1 & 0xFF) << 8) | ((b2 & 0xFF) << 16) | ((b3 & 0xFF) << 24);
}

__global__ __launch_bounds__(256)
void k1_proj(const float* __restrict__ x, const float* __restrict__ w1,
             const float* __restrict__ g1, const float* __restrict__ b1,
             const float* __restrict__ m1, const float* __restrict__ v1,
             signed char* __restrict__ s1)
{
    __shared__ signed char ldsA[128 * 128];
    __shared__ signed char ldsBT[32 * 128];
    __shared__ signed char ldsS[128 * 32];
    __shared__ float sInv[32], sBB[32];

    const int t = threadIdx.x;
    const int pos0 = blockIdx.x * 128;
    const int b = pos0 >> 15;
    const int p0 = pos0 & 32767;
    const float* xb = x + ((size_t)b * 128) * 32768 + p0;

    #pragma unroll
    for (int it = 0; it < 16; ++it) {
        int i = (it * 256 + t) * 4;
        int c = i >> 7, m = i & 127;
        const float4 xv = *(const float4*)(xb + (size_t)c * 32768 + m);
        ldsA[(m + 0) * 128 + c] = (signed char)sgnb(xv.x);
        ldsA[(m + 1) * 128 + c] = (signed char)sgnb(xv.y);
        ldsA[(m + 2) * 128 + c] = (signed char)sgnb(xv.z);
        ldsA[(m + 3) * 128 + c] = (signed char)sgnb(xv.w);
    }
    #pragma unroll
    for (int it = 0; it < 4; ++it) {
        int i4 = it * 256 + t;
        const float4 wv = *(const float4*)(w1 + (size_t)i4 * 4);
        ((int*)ldsBT)[i4] = pack4(sgnb(wv.x), sgnb(wv.y), sgnb(wv.z), sgnb(wv.w));
    }
    if (t < 32) {
        float inv = g1[t] / sqrtf(v1[t] + EPS);
        sInv[t] = inv; sBB[t] = b1[t] - m1[t] * inv;
    }
    __syncthreads();

    const int wave = t >> 5, lane = t & 31;
    const int hi = lane >> 4, lo = lane & 15;
    const int mRow = wave * 16 + lo;

    v8i acc0 = {}, acc1 = {};
    #pragma unroll
    for (int kc = 0; kc < 2; ++kc) {
        v8i a, bt0, bt1;
        #pragma unroll
        for (int v = 0; v < 8; ++v) {
            int ka = kc * 64 + (v >> 1) * 16 + hi * 8 + (v & 1) * 4;
            a[v] = *(const int*)(ldsA + mRow * 128 + ka);
            int kb = kc * 64 + (v >> 1) * 16 + hi * 8 + (v & 1) * 4;
            bt0[v] = *(const int*)(ldsBT + lo * 128 + kb);
            bt1[v] = *(const int*)(ldsBT + (16 + lo) * 128 + kb);
        }
        acc0 = __builtin_amdgcn_wmma_i32_16x16x64_iu8(true, a, true, bt0, acc0, false, false);
        acc1 = __builtin_amdgcn_wmma_i32_16x16x64_iu8(true, a, true, bt1, acc1, false, false);
    }

    #pragma unroll
    for (int j = 0; j < 8; ++j) {
        int M = wave * 16 + j + hi * 8;
        float z0 = (float)acc0[j] * sInv[lo] + sBB[lo];
        ldsS[M * 32 + lo] = (z0 > 0.0f) ? 1 : -1;
        float z1 = (float)acc1[j] * sInv[16 + lo] + sBB[16 + lo];
        ldsS[M * 32 + 16 + lo] = (z1 > 0.0f) ? 1 : -1;
    }
    __syncthreads();
    ((int4*)(s1 + (size_t)pos0 * 32))[t] = ((const int4*)ldsS)[t];
}

__global__ __launch_bounds__(256)
void k2_conv3x3(const signed char* __restrict__ s1, const float* __restrict__ w2,
                const float* __restrict__ g2, const float* __restrict__ b2,
                const float* __restrict__ m2, const float* __restrict__ v2,
                signed char* __restrict__ s2)
{
    __shared__ signed char ldsH[3 * 130 * 32];
    __shared__ signed char ldsBT[32 * 320];
    __shared__ signed char ldsS[128 * 32];
    __shared__ float sInv[32], sBB[32];

    const int t = threadIdx.x;
    const int blk = blockIdx.x;
    const int wh = blk & 1;
    const int h  = (blk >> 1) & 127;
    const int b  = blk >> 8;
    const int w0 = wh * 128;

    for (int i16 = t; i16 < 780; i16 += 256) {
        int c16 = (i16 & 1) * 16;
        int q  = i16 >> 1;
        int wl = q % 130, r = q / 130;
        int hh = h + r - 1, ww = w0 + wl - 1;
        int4 val = make_int4(0, 0, 0, 0);
        if (hh >= 0 && hh < 128 && ww >= 0 && ww < 256)
            val = *(const int4*)(s1 + (size_t)(b * 32768 + hh * 256 + ww) * 32 + c16);
        *(int4*)(ldsH + i16 * 16) = val;
    }
    for (int i4 = t; i4 < 2560; i4 += 256) {
        int n = i4 / 80, k4 = (i4 % 80) * 4;
        int val = 0;
        if (k4 < 288) {
            int tap = k4 >> 5, c = k4 & 31;
            const float* wp = w2 + (size_t)(n * 32 + c) * 9 + tap;
            val = pack4(sgnb(wp[0]), sgnb(wp[9]), sgnb(wp[18]), sgnb(wp[27]));
        }
        *(int*)(ldsBT + i4 * 4) = val;
    }
    if (t < 32) {
        float inv = g2[t] / sqrtf(v2[t] + EPS);
        sInv[t] = inv; sBB[t] = b2[t] - m2[t] * inv;
    }
    __syncthreads();

    const int wave = t >> 5, lane = t & 31;
    const int hi = lane >> 4, lo = lane & 15;
    const int wl = wave * 16 + lo;

    v8i acc0 = {}, acc1 = {};
    #pragma unroll
    for (int kc = 0; kc < 5; ++kc) {
        v8i a, bt0, bt1;
        #pragma unroll
        for (int v = 0; v < 8; ++v) {
            int ka = kc * 64 + (v >> 1) * 16 + hi * 8 + (v & 1) * 4;
            int av = 0;
            if (ka < 288) {
                int tap = ka >> 5, c = ka & 31;
                int ky = tap / 3, kx = tap - ky * 3;
                av = *(const int*)(ldsH + (ky * 130 + wl + kx) * 32 + c);
            }
            a[v] = av;
            int kb = kc * 64 + (v >> 1) * 16 + hi * 8 + (v & 1) * 4;
            bt0[v] = *(const int*)(ldsBT + lo * 320 + kb);
            bt1[v] = *(const int*)(ldsBT + (16 + lo) * 320 + kb);
        }
        acc0 = __builtin_amdgcn_wmma_i32_16x16x64_iu8(true, a, true, bt0, acc0, false, false);
        acc1 = __builtin_amdgcn_wmma_i32_16x16x64_iu8(true, a, true, bt1, acc1, false, false);
    }

    #pragma unroll
    for (int j = 0; j < 8; ++j) {
        int M = wave * 16 + j + hi * 8;
        float z0 = (float)acc0[j] * sInv[lo] + sBB[lo];
        ldsS[M * 32 + lo] = (z0 > 0.0f) ? 1 : -1;
        float z1 = (float)acc1[j] * sInv[16 + lo] + sBB[16 + lo];
        ldsS[M * 32 + 16 + lo] = (z1 > 0.0f) ? 1 : -1;
    }
    __syncthreads();
    ((int4*)(s2 + (size_t)(b * 32768 + h * 256 + w0) * 32))[t] = ((const int4*)ldsS)[t];
}

__global__ __launch_bounds__(256)
void k3_expand(const signed char* __restrict__ s2, const float* __restrict__ w3,
               const float* __restrict__ g3, const float* __restrict__ b3,
               const float* __restrict__ m3, const float* __restrict__ v3,
               const float* __restrict__ a3p, const float* __restrict__ aop,
               const float* __restrict__ x, float* __restrict__ out)
{
    __shared__ signed char ldsA[128 * 32];
    __shared__ signed char ldsBT[128 * 32];
    __shared__ float sInv[128], sBB[128];
    __shared__ float ldsHf[64 * 128];

    const int t = threadIdx.x;
    const int pos0 = blockIdx.x * 128;
    const int b = pos0 >> 15;
    const int p0 = pos0 & 32767;

    ((int4*)ldsA)[t] = ((const int4*)(s2 + (size_t)pos0 * 32))[t];
    #pragma unroll
    for (int it = 0; it < 4; ++it) {
        int i4 = it * 256 + t;
        const float4 wv = *(const float4*)(w3 + (size_t)i4 * 4);
        ((int*)ldsBT)[i4] = pack4(sgnb(wv.x), sgnb(wv.y), sgnb(wv.z), sgnb(wv.w));
    }
    if (t < 128) {
        float inv = g3[t] / sqrtf(v3[t] + EPS);
        sInv[t] = inv; sBB[t] = b3[t] - m3[t] * inv;
    }
    __syncthreads();

    const float a3v = a3p[0], aov = aop[0];
    const int wave = t >> 5, lane = t & 31;
    const int hi = lane >> 4, lo = lane & 15;
    const int mRow = wave * 16 + lo;

    v8i a;
    #pragma unroll
    for (int v = 0; v < 8; ++v) {
        int ka = (v >> 1) * 16 + hi * 8 + (v & 1) * 4;
        a[v] = (v < 4) ? *(const int*)(ldsA + mRow * 32 + ka) : 0;
    }

    v8i acc[8];
    v8i cz = {};
    #pragma unroll
    for (int nt = 0; nt < 8; ++nt) {
        v8i bt;
        #pragma unroll
        for (int v = 0; v < 8; ++v) {
            int kb = (v >> 1) * 16 + hi * 8 + (v & 1) * 4;
            bt[v] = (v < 4) ? *(const int*)(ldsBT + (nt * 16 + lo) * 32 + kb) : 0;
        }
        acc[nt] = __builtin_amdgcn_wmma_i32_16x16x64_iu8(true, a, true, bt, cz, false, false);
    }

    const float* xb = x + ((size_t)b * 128) * 32768 + p0;
    float* ob = out + ((size_t)b * 128) * 32768 + p0;

    #pragma unroll
    for (int half = 0; half < 2; ++half) {
        #pragma unroll
        for (int q = 0; q < 4; ++q) {
            int nt = half * 4 + q;
            int ch = nt * 16 + lo;
            float inv = sInv[ch], bb = sBB[ch];
            #pragma unroll
            for (int j = 0; j < 8; ++j) {
                int M = j + hi * 8;
                float z = (float)acc[nt][j] * inv + bb;
                ldsHf[(q * 16 + lo) * 128 + wave * 16 + M] = (z >= 0.0f) ? z : a3v * z;
            }
        }
        __syncthreads();
        #pragma unroll
        for (int it = 0; it < 8; ++it) {
            int e = (it * 256 + t) * 4;
            int c = e >> 7, m = e & 127;
            int ch = half * 64 + c;
            const float4 xv = *(const float4*)(xb + (size_t)ch * 32768 + m);
            const float4 hv = *(const float4*)(ldsHf + c * 128 + m);
            float4 o;
            float s0 = hv.x + xv.x; o.x = (s0 >= 0.0f) ? s0 : aov * s0;
            float s1 = hv.y + xv.y; o.y = (s1 >= 0.0f) ? s1 : aov * s1;
            float s2v = hv.z + xv.z; o.z = (s2v >= 0.0f) ? s2v : aov * s2v;
            float s3 = hv.w + xv.w; o.w = (s3 >= 0.0f) ? s3 : aov * s3;
            *(float4*)(ob + (size_t)ch * 32768 + m) = o;
        }
        __syncthreads();
    }
}

extern "C" void kernel_launch(void* const* d_in, const int* in_sizes, int n_in,
                              void* d_out, int out_size, void* d_ws, size_t ws_size,
                              hipStream_t stream)
{
    const float* x  = (const float*)d_in[0];
    const float* w1 = (const float*)d_in[1];
    const float* g1 = (const float*)d_in[2];
    const float* b1 = (const float*)d_in[3];
    const float* m1 = (const float*)d_in[4];
    const float* v1 = (const float*)d_in[5];
    const float* w2 = (const float*)d_in[7];
    const float* g2 = (const float*)d_in[8];
    const float* b2 = (const float*)d_in[9];
    const float* m2 = (const float*)d_in[10];
    const float* v2 = (const float*)d_in[11];
    const float* w3 = (const float*)d_in[13];
    const float* g3 = (const float*)d_in[14];
    const float* b3 = (const float*)d_in[15];
    const float* m3 = (const float*)d_in[16];
    const float* v3 = (const float*)d_in[17];
    const float* a3 = (const float*)d_in[18];
    const float* ao = (const float*)d_in[19];
    float* out = (float*)d_out;

    signed char* s1 = (signed char*)d_ws;
    signed char* s2 = s1 + (size_t)262144 * 32;

    k1_proj   <<<2048, 256, 0, stream>>>(x,  w1, g1, b1, m1, v1, s1);
    k2_conv3x3<<<2048, 256, 0, stream>>>(s1, w2, g2, b2, m2, v2, s2);
    k3_expand <<<2048, 256, 0, stream>>>(s2, w3, g3, b3, m3, v3, a3, ao, x, out);
}
